// MultibranchModel_58360015618167
// MI455X (gfx1250) — matrix operand order repaired and checked by reading, not compiled, not run on hardware
//
#include <hip/hip_runtime.h>
#include <hip/hip_bf16.h>
#include <math.h>

// ---- model dims ----
#define BB 8
#define LL 2048
#define JJ 25
#define FF 6
#define EE 64
#define NLAYERS 2
#define SLOPE 0.01f
#define CHUNKS 32
#define CLEN 64            // CHUNKS*CLEN == LL

typedef __attribute__((ext_vector_type(16))) _Float16 v16h;
typedef __attribute__((ext_vector_type(8)))  float    v8f;
typedef __attribute__((ext_vector_type(4)))  unsigned v4u;
typedef __attribute__((ext_vector_type(8)))  unsigned v8u;

// ---------------- WMMA fragment gathers (contiguous k order k = 16*(lane/16) + i, same in A and B) ----------------
__device__ __forceinline__ v16h load_a_frag(const _Float16* ldsA, int lane, int k0) {
    int m = lane & 15, g = lane >> 4;
    v16h a;
#pragma unroll
    for (int i = 0; i < 16; ++i) {
        int k = i + 16 * g + k0;
        a[i] = ldsA[m * 64 + k];
    }
    return a;
}
// B from weight stored row-major (N_out, K): y = x @ W^T  ->  B[k][n] = W[n][k]
__device__ __forceinline__ v16h load_b_fragT(const _Float16* ldsW, int lane, int k0, int n0) {
    int n = (lane & 15) + n0, g = lane >> 4;
    v16h b;
#pragma unroll
    for (int i = 0; i < 16; ++i) {
        int k = i + 16 * g + k0;
        b[i] = ldsW[n * 64 + k];
    }
    return b;
}
// B from weight stored row-major (K, N_out): y = x @ W  ->  B[k][n] = W[k][n]
__device__ __forceinline__ v16h load_b_frag(const _Float16* ldsW, int lane, int k0, int n0) {
    int n = (lane & 15) + n0, g = lane >> 4;
    v16h b;
#pragma unroll
    for (int i = 0; i < 16; ++i) {
        int k = i + 16 * g + k0;
        b[i] = ldsW[k * 64 + n];
    }
    return b;
}

__device__ __forceinline__ float leaky(float x) { return x > 0.f ? x : SLOPE * x; }
__device__ __forceinline__ float gelu_tanh(float x) {
    float t = x + 0.044715f * x * x * x;
    return 0.5f * x * (1.f + tanhf(0.7978845608028654f * t));
}

// ---------------- 1) input projection ----------------
__global__ void k_input_proj(const float* __restrict__ x, const float* __restrict__ W,
                             const float* __restrict__ b, float* __restrict__ h, int total) {
    int i = blockIdx.x * blockDim.x + threadIdx.x;
    if (i >= total) return;
    int e = i & 63;
    size_t row = (size_t)(i >> 6);
    const float* xr = x + row * FF;
    float acc = b[e];
#pragma unroll
    for (int f = 0; f < FF; ++f) acc += xr[f] * W[f * EE + e];
    h[i] = leaky(acc);
}

// ---------------- 2) LayerNorm over E (one wave per row) -> f16 ----------------
__global__ void k_ln(const float* __restrict__ x, _Float16* __restrict__ out,
                     const float* __restrict__ sc, const float* __restrict__ bi, int per_joint) {
    int wave = threadIdx.x >> 5, lane = threadIdx.x & 31;
    size_t row = (size_t)blockIdx.x * 8 + wave;
    const float* xr = x + row * EE;
    float a0 = xr[lane * 2], a1 = xr[lane * 2 + 1];
    float s = a0 + a1;
#pragma unroll
    for (int o = 16; o > 0; o >>= 1) s += __shfl_xor(s, o, 32);
    float mean = s * (1.f / 64.f);
    float d0 = a0 - mean, d1 = a1 - mean;
    float vv = d0 * d0 + d1 * d1;
#pragma unroll
    for (int o = 16; o > 0; o >>= 1) vv += __shfl_xor(vv, o, 32);
    float inv = rsqrtf(vv * (1.f / 64.f) + 1e-5f);
    int base = per_joint ? ((int)(row % JJ)) * EE : 0;
    out[row * EE + lane * 2]     = (_Float16)(d0 * inv * sc[base + lane * 2]     + bi[base + lane * 2]);
    out[row * EE + lane * 2 + 1] = (_Float16)(d1 * inv * sc[base + lane * 2 + 1] + bi[base + lane * 2 + 1]);
}

// ---------------- 3) Bu = (xn @ Bre^T + i xn @ Bim^T) * gamma   (per joint, WMMA) ----------------
__global__ void k_gemm_bu(const _Float16* __restrict__ xn16,
                          const float* __restrict__ Bre, const float* __restrict__ Bim,
                          const float* __restrict__ gamma_log,
                          float* __restrict__ bu_re, float* __restrict__ bu_im) {
    int j = blockIdx.y;
    __shared__ _Float16 wre[4096], wim[4096];
    __shared__ _Float16 atile[4][1024];
    for (int idx = threadIdx.x; idx < 4096; idx += blockDim.x) {
        wre[idx] = (_Float16)Bre[j * 4096 + idx];
        wim[idx] = (_Float16)Bim[j * 4096 + idx];
    }
    int wave = threadIdx.x >> 5, lane = threadIdx.x & 31;
    int row0 = (blockIdx.x * 4 + wave) * 16;
    _Float16* A = atile[wave];
    for (int i = lane; i < 1024; i += 32) {
        int r = i >> 6, e = i & 63;
        A[i] = xn16[((size_t)(row0 + r) * JJ + j) * EE + e];
    }
    __syncthreads();
#pragma unroll
    for (int nt = 0; nt < 4; ++nt) {
        v8f cre = {}, cim = {};
#pragma unroll
        for (int kc = 0; kc < 2; ++kc) {
            v16h a  = load_a_frag(A, lane, kc * 32);
            v16h br = load_b_fragT(wre, lane, kc * 32, nt * 16);
            v16h bi = load_b_fragT(wim, lane, kc * 32, nt * 16);
            cre = __builtin_amdgcn_wmma_f32_16x16x32_f16(false, a, false, br, (short)0, cre, false, false);
            cim = __builtin_amdgcn_wmma_f32_16x16x32_f16(false, a, false, bi, (short)0, cim, false, false);
        }
        int g = lane >> 4, n = (lane & 15) + nt * 16;
        float gam = expf(gamma_log[j * EE + n]);
#pragma unroll
        for (int r = 0; r < 8; ++r) {
            int m = r + 8 * g;
            size_t o = ((size_t)(row0 + m) * JJ + j) * EE + n;
            bu_re[o] = cre[r] * gam;
            bu_im[o] = cim[r] * gam;
        }
    }
}

// ---------------- 4a) chunked local scans (phase A) ----------------
// thread per (channel, chunk): 64-step local complex scan in place, emit carry
__global__ void k_scan_local(float* __restrict__ bu_re, float* __restrict__ bu_im,
                             const float* __restrict__ nu_log, const float* __restrict__ th_log,
                             float* __restrict__ carry_re, float* __restrict__ carry_im) {
    int t = blockIdx.x * blockDim.x + threadIdx.x;            // B*J*E*CHUNKS = 409600
    if (t >= BB * JJ * EE * CHUNKS) return;
    int chunk = t & (CHUNKS - 1);
    int c = t >> 5;                                           // channel = (b*J+j)*E+n
    int n = c & 63, j = (c >> 6) % JJ, b = c / (JJ * EE);
    float enu = expf(nu_log[j * EE + n]);
    float eth = expf(th_log[j * EE + n]);
    float mag = expf(-enu);
    float lre = mag * cosf(eth), lim = mag * sinf(eth);
    const size_t stride = (size_t)JJ * EE;
    size_t idx = ((size_t)b * LL + chunk * CLEN) * stride + (size_t)j * EE + n;
    float hre = 0.f, him = 0.f;
    for (int s = 0; s < CLEN; ++s) {
        __builtin_prefetch(&bu_re[idx + stride], 0, 0);       // global_prefetch_b8 (speculative)
        __builtin_prefetch(&bu_im[idx + stride], 0, 0);
        float bre = bu_re[idx], bim = bu_im[idx];
        float nr = lre * hre - lim * him + bre;
        float ni = lre * him + lim * hre + bim;
        hre = nr; him = ni;
        bu_re[idx] = hre; bu_im[idx] = him;
        idx += stride;
    }
    carry_re[t] = hre; carry_im[t] = him;
}

// ---------------- 4b) scan across chunk carries (phase B) ----------------
__global__ void k_scan_carry(float* __restrict__ carry_re, float* __restrict__ carry_im,
                             const float* __restrict__ nu_log, const float* __restrict__ th_log) {
    int c = blockIdx.x * blockDim.x + threadIdx.x;            // B*J*E = 12800
    if (c >= BB * JJ * EE) return;
    int n = c & 63, j = (c >> 6) % JJ;
    float enu = expf(nu_log[j * EE + n]);
    float eth = expf(th_log[j * EE + n]);
    float m64 = expf(-(float)CLEN * enu);                     // |lam|^64
    float a64 = (float)CLEN * eth;
    float l64re = m64 * cosf(a64), l64im = m64 * sinf(a64);
    float sre = 0.f, sim = 0.f;
    int base = c * CHUNKS;
    for (int k = 0; k < CHUNKS; ++k) {
        float fr = carry_re[base + k], fi = carry_im[base + k];
        carry_re[base + k] = sre;                             // replace with carry-IN of chunk k
        carry_im[base + k] = sim;
        float nr = l64re * sre - l64im * sim + fr;
        float ni = l64re * sim + l64im * sre + fi;
        sre = nr; sim = ni;
    }
}

// ---------------- 4c) elementwise correction (phase C, fully parallel) ----------------
// h_t = local_t + lam^(t_local+1) * carry_in(chunk);  lam^m computed in closed form
__global__ void k_scan_apply(float* __restrict__ bu_re, float* __restrict__ bu_im,
                             const float* __restrict__ nu_log, const float* __restrict__ th_log,
                             const float* __restrict__ carry_re, const float* __restrict__ carry_im,
                             int total) {
    int i = blockIdx.x * blockDim.x + threadIdx.x;            // B*L*J*E
    if (i >= total) return;
    int n = i & 63;
    int rj = i >> 6;                                          // (b*L+l)*J + j
    int j = rj % JJ;
    int row = rj / JJ;                                        // b*L + l
    int l = row & (LL - 1), b = row >> 11;
    int chunk = l >> 6, tl = l & 63;
    int cidx = (((b * JJ + j) * EE) + n) * CHUNKS + chunk;
    float sre = carry_re[cidx], sim = carry_im[cidx];
    float enu = expf(nu_log[j * EE + n]);
    float eth = expf(th_log[j * EE + n]);
    float m = (float)(tl + 1);
    float pm = expf(-m * enu);
    float pa = m * eth;
    float pre = pm * cosf(pa), pim = pm * sinf(pa);
    bu_re[i] += pre * sre - pim * sim;
    bu_im[i] += pre * sim + pim * sre;
}

// ---------------- 5) y = hs_re@Cre^T - hs_im@Cim^T + D*xn ; h += gelu(y)  (WMMA) ----------------
__global__ void k_gemm_c(const float* __restrict__ hs_re, const float* __restrict__ hs_im,
                         const float* __restrict__ Cre, const float* __restrict__ Cim,
                         const _Float16* __restrict__ xn16, const float* __restrict__ Dm,
                         float* __restrict__ h) {
    int j = blockIdx.y;
    __shared__ _Float16 wre[4096], wim[4096];                 // wim holds -Cim
    __shared__ _Float16 are[4][1024], aim[4][1024];
    for (int idx = threadIdx.x; idx < 4096; idx += blockDim.x) {
        wre[idx] = (_Float16)Cre[j * 4096 + idx];
        wim[idx] = (_Float16)(-Cim[j * 4096 + idx]);
    }
    int wave = threadIdx.x >> 5, lane = threadIdx.x & 31;
    int row0 = (blockIdx.x * 4 + wave) * 16;
    for (int i = lane; i < 1024; i += 32) {
        int r = i >> 6, e = i & 63;
        size_t o = ((size_t)(row0 + r) * JJ + j) * EE + e;
        are[wave][i] = (_Float16)hs_re[o];
        aim[wave][i] = (_Float16)hs_im[o];
    }
    __syncthreads();
#pragma unroll
    for (int nt = 0; nt < 4; ++nt) {
        v8f c = {};
#pragma unroll
        for (int kc = 0; kc < 2; ++kc) {
            v16h a1 = load_a_frag(are[wave], lane, kc * 32);
            v16h b1 = load_b_fragT(wre, lane, kc * 32, nt * 16);
            c = __builtin_amdgcn_wmma_f32_16x16x32_f16(false, a1, false, b1, (short)0, c, false, false);
            v16h a2 = load_a_frag(aim[wave], lane, kc * 32);
            v16h b2 = load_b_fragT(wim, lane, kc * 32, nt * 16);
            c = __builtin_amdgcn_wmma_f32_16x16x32_f16(false, a2, false, b2, (short)0, c, false, false);
        }
        int g = lane >> 4, n = (lane & 15) + nt * 16;
        float dcoef = Dm[j * EE + n];
#pragma unroll
        for (int r = 0; r < 8; ++r) {
            int m = r + 8 * g;
            size_t o = ((size_t)(row0 + m) * JJ + j) * EE + n;
            float y = c[r] + dcoef * (float)xn16[o];
            h[o] += gelu_tanh(y);
        }
    }
}

// ---------------- 6) fused K/Q/V projections (WMMA, TDM-staged A tile) ----------------
// rows = flattened (b,l,j) = 409600; block = 128 (4 waves * 16 rows, 64 rows contiguous)
__global__ void k_gemm_kqv(const _Float16* __restrict__ hn16,
                           const float* __restrict__ Wk, const float* __restrict__ bk,
                           const float* __restrict__ Wq, const float* __restrict__ bq,
                           const float* __restrict__ Wv, const float* __restrict__ bv,
                           float* __restrict__ Kd, float* __restrict__ Qd, float* __restrict__ Vd) {
    __shared__ _Float16 wk[4096], wq[4096], wv[4096];
    __shared__ _Float16 atile[4][1024];
    for (int idx = threadIdx.x; idx < 4096; idx += blockDim.x) {
        wk[idx] = (_Float16)Wk[idx];
        wq[idx] = (_Float16)Wq[idx];
        wv[idx] = (_Float16)Wv[idx];
    }
    // --- TDM: one wave issues an async tensor DMA of the block's 64 contiguous
    //     rows (64*64 f16 = 8KB) from global into LDS, then waits on TENSORcnt.
    if (threadIdx.x < 32) {
        unsigned lds_off = (unsigned)(uintptr_t)(&atile[0][0]);          // LDS aperture: low 32 bits
        unsigned long long ga =
            (unsigned long long)(uintptr_t)(hn16 + (size_t)blockIdx.x * 64 * EE);
        v4u g0;
        g0[0] = 1u;                                   // count=1, is_restore=0, gather off
        g0[1] = lds_off;                              // lds_addr
        g0[2] = (unsigned)ga;                         // global_addr[31:0]
        g0[3] = (unsigned)((ga >> 32) & 0x1FFFFFFu) | (2u << 30);  // addr[56:32] | type=2
        v8u g1;
        g1[0] = (1u << 16);                           // workgroup_mask=0, data_size=1 (2B)
        g1[1] = (4096u << 16);                        // tensor_dim0[15:0] = 4096
        g1[2] = 0u;                                   // tensor_dim0 hi, tensor_dim1[15:0]=0
        g1[3] = (4096u << 16);                        // tile_dim0 = 4096 (1-D tile)
        g1[4] = 0u;                                   // tile_dim1=0, tile_dim2=0
        g1[5] = 4096u;                                // tensor_dim0_stride lo
        g1[6] = 0u;
        g1[7] = 0u;
        asm volatile("tensor_load_to_lds %0, %1" :: "s"(g0), "s"(g1) : "memory");
        __builtin_amdgcn_s_wait_tensorcnt(0);
    }
    __syncthreads();

    int wave = threadIdx.x >> 5, lane = threadIdx.x & 31;
    int row0 = (blockIdx.x * 4 + wave) * 16;
    _Float16* A = atile[wave];
#pragma unroll
    for (int nt = 0; nt < 4; ++nt) {
        v8f ck = {}, cq = {}, cv = {};
#pragma unroll
        for (int kc = 0; kc < 2; ++kc) {
            v16h a = load_a_frag(A, lane, kc * 32);
            v16h fk = load_b_frag(wk, lane, kc * 32, nt * 16);
            v16h fq = load_b_frag(wq, lane, kc * 32, nt * 16);
            v16h fv = load_b_frag(wv, lane, kc * 32, nt * 16);
            ck = __builtin_amdgcn_wmma_f32_16x16x32_f16(false, a, false, fk, (short)0, ck, false, false);
            cq = __builtin_amdgcn_wmma_f32_16x16x32_f16(false, a, false, fq, (short)0, cq, false, false);
            cv = __builtin_amdgcn_wmma_f32_16x16x32_f16(false, a, false, fv, (short)0, cv, false, false);
        }
        int g = lane >> 4, n = (lane & 15) + nt * 16;
#pragma unroll
        for (int r = 0; r < 8; ++r) {
            int m = r + 8 * g;
            size_t o = (size_t)(row0 + m) * EE + n;
            Kd[o] = ck[r] + bk[n];
            Qd[o] = cq[r] + bq[n];
            Vd[o] = cv[r] + bv[n];
        }
    }
}

// ---------------- 7) 25x25 joint attention per (b,l) ----------------
__global__ void k_attn(const float* __restrict__ Kd, const float* __restrict__ Qd,
                       const float* __restrict__ Vd, float* __restrict__ h) {
    __shared__ float sQ[JJ * EE], sK[JJ * EE], sV[JJ * EE];
    size_t base = (size_t)blockIdx.x * JJ * EE;
    int lane = threadIdx.x;
    for (int i = lane; i < JJ * EE; i += 32) {
        sQ[i] = Qd[base + i]; sK[i] = Kd[base + i]; sV[i] = Vd[base + i];
    }
    __syncthreads();
    if (lane < JJ) {
        float s[JJ];
        float mx = -1e30f;
#pragma unroll
        for (int j = 0; j < JJ; ++j) {
            float d = 0.f;
#pragma unroll
            for (int e = 0; e < EE; ++e) d += sQ[lane * EE + e] * sK[j * EE + e];
            s[j] = d;
            mx = fmaxf(mx, d);
        }
        float sum = 0.f;
#pragma unroll
        for (int j = 0; j < JJ; ++j) { s[j] = expf(s[j] - mx); sum += s[j]; }
        float inv = 1.f / sum;
        for (int e = 0; e < EE; ++e) {
            float acc = 0.f;
#pragma unroll
            for (int j = 0; j < JJ; ++j) acc += s[j] * sV[j * EE + e];
            size_t o = base + (size_t)lane * EE + e;
            h[o] = leaky(acc * inv) + h[o];
        }
    }
}

// ---------------- 8) pooling: stage 1 (per 32-l chunk), stage 2 (combine) ----------------
__global__ void k_pool1(const float* __restrict__ h, float* __restrict__ partial) {
    int b = blockIdx.x >> 6, chunk = blockIdx.x & 63;         // 64 chunks of 32 l's
    int e = threadIdx.x;
    float acc = 0.f;
    for (int li = 0; li < 32; ++li) {
        int l = chunk * 32 + li;
        size_t base = ((size_t)b * LL + l) * JJ * EE + e;
        float mx = -1e30f;
#pragma unroll
        for (int j = 0; j < JJ; ++j) mx = fmaxf(mx, h[base + (size_t)j * EE]);
        acc += mx;
    }
    partial[(size_t)blockIdx.x * EE + e] = acc;               // [b][chunk][e]
}
__global__ void k_pool2(const float* __restrict__ partial, float* __restrict__ pooled) {
    int i = blockIdx.x * blockDim.x + threadIdx.x;            // B*E = 512
    if (i >= BB * EE) return;
    int e = i & 63, b = i >> 6;
    float acc = 0.f;
    for (int k = 0; k < 64; ++k) acc += partial[(size_t)(b * 64 + k) * EE + e];
    pooled[i] = acc * (1.f / (float)LL);
}

// ---------------- 9) head ----------------
__global__ void k_head(const float* __restrict__ pooled, const float* __restrict__ W1,
                       const float* __restrict__ b1, const float* __restrict__ W2,
                       const float* __restrict__ b2, float* __restrict__ out) {
    __shared__ float red[EE];
    int b = blockIdx.x, e = threadIdx.x;
    float acc = b1[e];
#pragma unroll
    for (int k = 0; k < EE; ++k) acc += pooled[b * EE + k] * W1[k * EE + e];
    red[e] = leaky(acc) * W2[e];
    __syncthreads();
    for (int o = 32; o > 0; o >>= 1) {
        if (e < o) red[e] += red[e + o];
        __syncthreads();
    }
    if (e == 0) out[b] = 1.f / (1.f + expf(-(red[0] + b2[0])));
}

// ---------------- orchestration ----------------
extern "C" void kernel_launch(void* const* d_in, const int* in_sizes, int n_in,
                              void* d_out, int out_size, void* d_ws, size_t ws_size,
                              hipStream_t stream) {
    const float* x          = (const float*)d_in[0];
    const float* W_in       = (const float*)d_in[1];
    const float* b_in       = (const float*)d_in[2];
    const float* ln_t_scale = (const float*)d_in[3];
    const float* ln_t_bias  = (const float*)d_in[4];
    const float* nu_log     = (const float*)d_in[5];
    const float* theta_log  = (const float*)d_in[6];
    const float* gamma_log  = (const float*)d_in[7];
    const float* B_re       = (const float*)d_in[8];
    const float* B_im       = (const float*)d_in[9];
    const float* C_re       = (const float*)d_in[10];
    const float* C_im       = (const float*)d_in[11];
    const float* Dp         = (const float*)d_in[12];
    const float* ln_g_scale = (const float*)d_in[13];
    const float* ln_g_bias  = (const float*)d_in[14];
    const float* Wk         = (const float*)d_in[15];
    const float* bk         = (const float*)d_in[16];
    const float* Wq         = (const float*)d_in[17];
    const float* bq         = (const float*)d_in[18];
    const float* Wv         = (const float*)d_in[19];
    const float* bv         = (const float*)d_in[20];
    const float* W1         = (const float*)d_in[21];
    const float* b1         = (const float*)d_in[22];
    const float* W2         = (const float*)d_in[23];
    const float* b2         = (const float*)d_in[24];

    const size_t NTOT = (size_t)BB * LL * JJ * EE;            // 26,214,400
    char* ws = (char*)d_ws;
    float*     h        = (float*)ws;                         // 104,857,600 B
    _Float16*  xn16     = (_Float16*)(ws + NTOT * 4);         //  52,428,800 B
    float*     bu_re    = (float*)(ws + NTOT * 6);            // 104,857,600 B (later: K)
    float*     bu_im    = (float*)(ws + NTOT * 10);           // 104,857,600 B (later: Q)
    float*     Vbuf     = (float*)(ws + NTOT * 14);           // 104,857,600 B
    float*     pooled   = (float*)(ws + NTOT * 18);           //       2,048 B
    float*     carry_re = (float*)(ws + NTOT * 18 + 4096);    //   1,638,400 B
    float*     carry_im = (float*)(ws + NTOT * 18 + 4096 + 1638400);
    float*     partial  = (float*)(ws + NTOT * 18 + 4096 + 2 * 1638400); // 131,072 B

    const int total = (int)NTOT;
    k_input_proj<<<(total + 255) / 256, 256, 0, stream>>>(x, W_in, b_in, h, total);

    const int JE = JJ * EE;            // 1600
    const int JEE = JJ * EE * EE;      // 102400
    const int NCH = BB * JJ * EE;      // 12800 channels
    for (int l = 0; l < NLAYERS; ++l) {
        k_ln<<<(BB * LL * JJ) / 8, 256, 0, stream>>>(h, xn16,
            ln_t_scale + l * JE, ln_t_bias + l * JE, 1);
        k_gemm_bu<<<dim3(256, JJ), 128, 0, stream>>>(xn16,
            B_re + (size_t)l * JEE, B_im + (size_t)l * JEE, gamma_log + l * JE, bu_re, bu_im);
        k_scan_local<<<(NCH * CHUNKS + 255) / 256, 256, 0, stream>>>(bu_re, bu_im,
            nu_log + l * JE, theta_log + l * JE, carry_re, carry_im);
        k_scan_carry<<<(NCH + 255) / 256, 256, 0, stream>>>(carry_re, carry_im,
            nu_log + l * JE, theta_log + l * JE);
        k_scan_apply<<<(total + 255) / 256, 256, 0, stream>>>(bu_re, bu_im,
            nu_log + l * JE, theta_log + l * JE, carry_re, carry_im, total);
        k_gemm_c<<<dim3(256, JJ), 128, 0, stream>>>(bu_re, bu_im,
            C_re + (size_t)l * JEE, C_im + (size_t)l * JEE, xn16, Dp + l * JE, h);
    }

    k_ln<<<(BB * LL * JJ) / 8, 256, 0, stream>>>(h, xn16, ln_g_scale, ln_g_bias, 0);
    k_gemm_kqv<<<(BB * LL * JJ) / 64, 128, 0, stream>>>(xn16, Wk, bk, Wq, bq, Wv, bv,
                                                        bu_re, bu_im, Vbuf);
    k_attn<<<BB * LL, 32, 0, stream>>>(bu_re, bu_im, Vbuf, h);
    k_pool1<<<BB * 64, 64, 0, stream>>>(h, partial);
    k_pool2<<<2, 256, 0, stream>>>(partial, pooled);
    k_head<<<BB, EE, 0, stream>>>(pooled, W1, b1, W2, b2, (float*)d_out);
}
